// FixedUniformWeightGNN_38878043964036
// MI455X (gfx1250) — compile-verified, matrix operand order checked by reading, not run on hardware
//
#include <hip/hip_runtime.h>

typedef __attribute__((ext_vector_type(8)))  _Float16 v8h;
typedef __attribute__((ext_vector_type(16))) _Float16 v16h;
typedef __attribute__((ext_vector_type(8)))  float    v8f;
typedef __attribute__((ext_vector_type(4)))  int      v4i;

#define GDIM 128                 // feature dimension
#define WPB  8                   // waves per block
#define EPW  16                  // edges per wave (WMMA M)
#define WPAD 8                   // row padding in halfs (16B) -> conflict-free b128
#define WSTRIDE (GDIM + WPAD)    // 136 halfs = 272B row stride in LDS

#ifndef HAVE_ASYNC_LDS
#if defined(__has_builtin)
#if __has_builtin(__builtin_amdgcn_global_load_async_to_lds_b128)
#define HAVE_ASYNC_LDS 1
#endif
#endif
#endif
#ifndef HAVE_ASYNC_LDS
#define HAVE_ASYNC_LDS 0
#endif

// ---------------------------------------------------------------------------
// f32 -> f16 conversion (z, W1, W2 staged as f16 for WMMA)
// ---------------------------------------------------------------------------
__global__ void cvt_f32_to_f16_kernel(const float* __restrict__ in,
                                      _Float16* __restrict__ out, int n) {
  int i = blockIdx.x * blockDim.x + threadIdx.x;
  int stride = gridDim.x * blockDim.x;
  for (; i < n; i += stride) out[i] = (_Float16)in[i];
}

// ---------------------------------------------------------------------------
__global__ void degree_kernel(const int* __restrict__ src,
                              unsigned int* __restrict__ deg, int E) {
  int e = blockIdx.x * blockDim.x + threadIdx.x;
  if (e < E) atomicAdd(&deg[src[e]], 1u);
}

__global__ void alpha_kernel(const int* __restrict__ src,
                             const unsigned int* __restrict__ deg,
                             float* __restrict__ alpha, int E) {
  int e = blockIdx.x * blockDim.x + threadIdx.x;
  if (e < E) alpha[e] = 1.0f / ((float)deg[src[e]] + 1e-12f);
}

// ---------------------------------------------------------------------------
// build a 32B WMMA fragment from two 16B-aligned LDS chunks (padded rows are
// only 16B aligned, so force b128 pairs explicitly)
// ---------------------------------------------------------------------------
__device__ __forceinline__ v16h lds_frag(const _Float16* p) {
  v8h lo = *reinterpret_cast<const v8h*>(p);
  v8h hi = *reinterpret_cast<const v8h*>(p + 8);
  return __builtin_shufflevector(lo, hi, 0, 1, 2, 3, 4, 5, 6, 7,
                                 8, 9, 10, 11, 12, 13, 14, 15);
}

// stage one 128x128 f16 matrix into padded LDS (16B chunks; gfx1250 async
// global->LDS DMA when available, plain vector copy otherwise)
__device__ __forceinline__ void stage_weights(const _Float16* __restrict__ g,
                                              _Float16* s, int tid) {
  for (int i = tid; i < (GDIM * GDIM) / 8; i += 256) {   // 2048 chunks of 8 halfs
    int r = i >> 4, c8 = i & 15;
    const _Float16* gp = g + r * GDIM + c8 * 8;
    _Float16* sp = s + r * WSTRIDE + c8 * 8;
#if HAVE_ASYNC_LDS
    __builtin_amdgcn_global_load_async_to_lds_b128(
        (__attribute__((address_space(1))) v4i*)(_Float16*)gp,
        (__attribute__((address_space(3))) v4i*)sp, 0, 0);
#else
    *reinterpret_cast<uint4*>(sp) = *reinterpret_cast<const uint4*>(gp);
#endif
  }
}

__device__ __forceinline__ void wait_stage_done() {
#if HAVE_ASYNC_LDS
#if defined(__has_builtin) && __has_builtin(__builtin_amdgcn_s_wait_asynccnt)
  __builtin_amdgcn_s_wait_asynccnt(0);
#else
  asm volatile("s_wait_asynccnt 0x0" ::: "memory");
#endif
#endif
}

// ---------------------------------------------------------------------------
// Main edge kernel (grid-stride over 128-edge groups; weights staged once):
//   h   = relu(z[dst] @ W1^T + b1)        (WMMA f32_16x16x32_f16)
//   msg = (h @ W2^T + b2) * alpha[e]      (WMMA f32_16x16x32_f16)
//   agg[src[e]] += msg                     (global_atomic_add_f32)
// ---------------------------------------------------------------------------
__launch_bounds__(256)
__global__ void edge_mlp_kernel(const _Float16* __restrict__ zh,
                                const int* __restrict__ src,
                                const int* __restrict__ dst,
                                const float* __restrict__ alpha,
                                const _Float16* __restrict__ w1h,
                                const _Float16* __restrict__ w2h,
                                const float* __restrict__ b1,
                                const float* __restrict__ b2,
                                float* __restrict__ agg, int E, int nGroups) {
  __shared__ _Float16 sW1[GDIM * WSTRIDE];       // 34 KB, padded rows
  __shared__ _Float16 sW2[GDIM * WSTRIDE];       // 34 KB
  __shared__ _Float16 sH[WPB][EPW * WSTRIDE];    // 34 KB: per-wave h staging
  __shared__ float    sB1[GDIM], sB2[GDIM];
  __shared__ float    sAlpha[WPB][EPW];
  __shared__ int      sSrc[WPB][EPW];

  const int tid  = threadIdx.x;
  const int wave = tid >> 5;
  const int lane = tid & 31;
  const int row  = lane & 15;   // A row / B column / D column
  const int half = lane >> 4;   // K-half selector

  stage_weights(w1h, sW1, tid);
  stage_weights(w2h, sW2, tid);
  if (tid < GDIM) { sB1[tid] = b1[tid]; sB2[tid] = b2[tid]; }
  wait_stage_done();
  __syncthreads();

  for (int g = blockIdx.x; g < nGroups; g += gridDim.x) {
    const int edgeBase = (g * WPB + wave) * EPW;

    if (lane < EPW) {
      int e = edgeBase + lane;
      e = e < E ? e : E - 1;
      sSrc[wave][lane]   = src[e];
      sAlpha[wave][lane] = alpha[e];
    }

    // gather A fragments straight from global: lane owns (row, K-half);
    // 16 contiguous f16 per K-block, rows 256B-aligned -> clean b128 pairs
    int ge = edgeBase + row;
    ge = ge < E ? ge : E - 1;
    const _Float16* zrow = zh + (size_t)dst[ge] * GDIM;
    v16h a[4];
    #pragma unroll
    for (int kb = 0; kb < 4; ++kb)
      a[kb] = *reinterpret_cast<const v16h*>(zrow + kb * 32 + half * 16);

    // prefetch next group's gather rows (global_prefetch_b8) while WMMAs run
    int gn = g + gridDim.x;
    if (gn < nGroups) {
      int gen = (gn * WPB + wave) * EPW + row;
      gen = gen < E ? gen : E - 1;
      __builtin_prefetch(zh + (size_t)dst[gen] * GDIM + half * 64, 0, 1);
    }

    // ---- layer 1: h = relu(A @ W1^T + b1) -> per-wave LDS staging
    #pragma unroll
    for (int nt = 0; nt < 8; ++nt) {
      v8f c = {};
      #pragma unroll
      for (int kb = 0; kb < 4; ++kb) {
        // B[k][n] = W1[nt*16+n][k]; padded stride -> conflict-free b128
        v16h b = lds_frag(&sW1[(nt * 16 + row) * WSTRIDE + kb * 32 + half * 16]);
        c = __builtin_amdgcn_wmma_f32_16x16x32_f16(false, a[kb], false, b,
                                                   (short)0, c, false, false);
      }
      float bias = sB1[nt * 16 + row];
      #pragma unroll
      for (int r = 0; r < 8; ++r) {
        float h = c[r] + bias;
        h = h > 0.0f ? h : 0.0f;
        int mrow = r + 8 * half;               // D layout: M = r + 8*(lane>=16)
        sH[wave][mrow * WSTRIDE + nt * 16 + row] = (_Float16)h;
      }
    }
    // sH is per-wave; same-wave LDS ops are in-order (DScnt) -> no barrier

    // ---- layer 2: msg = (H @ W2^T + b2) * alpha, scatter-add to agg
    #pragma unroll
    for (int nt = 0; nt < 8; ++nt) {
      v8f c = {};
      #pragma unroll
      for (int kb = 0; kb < 4; ++kb) {
        v16h ah = lds_frag(&sH[wave][row * WSTRIDE + kb * 32 + half * 16]);
        v16h b  = lds_frag(&sW2[(nt * 16 + row) * WSTRIDE + kb * 32 + half * 16]);
        c = __builtin_amdgcn_wmma_f32_16x16x32_f16(false, ah, false, b,
                                                   (short)0, c, false, false);
      }
      int col = nt * 16 + row;
      float bias = sB2[col];
      #pragma unroll
      for (int r = 0; r < 8; ++r) {
        int mrow = r + 8 * half;
        if (edgeBase + mrow < E) {
          float m = (c[r] + bias) * sAlpha[wave][mrow];
          atomicAdd(&agg[(size_t)sSrc[wave][mrow] * GDIM + col], m);
        }
      }
    }
  }
}

// ---------------------------------------------------------------------------
// residual + LayerNorm: one wave32 per node, 4 elems/lane, shfl reduction
// ---------------------------------------------------------------------------
__launch_bounds__(256)
__global__ void layernorm_kernel(const float* __restrict__ z,
                                 const float* __restrict__ agg,
                                 const float* __restrict__ gamma,
                                 const float* __restrict__ beta,
                                 float* __restrict__ out, int N) {
  const int wave = threadIdx.x >> 5;
  const int lane = threadIdx.x & 31;
  const int node = blockIdx.x * (blockDim.x >> 5) + wave;
  if (node >= N) return;

  float4 vz = reinterpret_cast<const float4*>(z   + (size_t)node * GDIM)[lane];
  float4 va = reinterpret_cast<const float4*>(agg + (size_t)node * GDIM)[lane];
  float x0 = vz.x + va.x, x1 = vz.y + va.y, x2 = vz.z + va.z, x3 = vz.w + va.w;

  float s  = x0 + x1 + x2 + x3;
  float ss = x0 * x0 + x1 * x1 + x2 * x2 + x3 * x3;
  #pragma unroll
  for (int off = 16; off > 0; off >>= 1) {
    s  += __shfl_xor(s,  off, 32);
    ss += __shfl_xor(ss, off, 32);
  }
  float mean = s * (1.0f / GDIM);
  float var  = ss * (1.0f / GDIM) - mean * mean;
  float rstd = rsqrtf(var + 1e-5f);

  float4 gm = reinterpret_cast<const float4*>(gamma)[lane];
  float4 bt = reinterpret_cast<const float4*>(beta)[lane];
  float4 o;
  o.x = (x0 - mean) * rstd * gm.x + bt.x;
  o.y = (x1 - mean) * rstd * gm.y + bt.y;
  o.z = (x2 - mean) * rstd * gm.z + bt.z;
  o.w = (x3 - mean) * rstd * gm.w + bt.w;
  reinterpret_cast<float4*>(out + (size_t)node * GDIM)[lane] = o;
}

// ---------------------------------------------------------------------------
extern "C" void kernel_launch(void* const* d_in, const int* in_sizes, int n_in,
                              void* d_out, int out_size, void* d_ws, size_t ws_size,
                              hipStream_t stream) {
  const float* z     = (const float*)d_in[0];
  const int*   ei    = (const int*)d_in[1];
  const float* W1    = (const float*)d_in[2];
  const float* b1    = (const float*)d_in[3];
  const float* W2    = (const float*)d_in[4];
  const float* b2    = (const float*)d_in[5];
  const float* gamma = (const float*)d_in[6];
  const float* beta  = (const float*)d_in[7];

  const int N = in_sizes[0] / GDIM;
  const int E = in_sizes[1] / 2;
  const int* src = ei;
  const int* dst = ei + E;

  float* out   = (float*)d_out;
  float* alpha = out + (size_t)N * GDIM;   // tuple output #2

  // workspace layout
  char* ws = (char*)d_ws;
  size_t off = 0;
  float* agg = (float*)(ws + off);               off += (size_t)N * GDIM * sizeof(float);
  unsigned int* deg = (unsigned int*)(ws + off); off += (size_t)N * sizeof(unsigned int);
  off = (off + 255) & ~(size_t)255;
  _Float16* zh  = (_Float16*)(ws + off);         off += (size_t)N * GDIM * sizeof(_Float16);
  _Float16* w1h = (_Float16*)(ws + off);         off += (size_t)GDIM * GDIM * sizeof(_Float16);
  _Float16* w2h = (_Float16*)(ws + off);         off += (size_t)GDIM * GDIM * sizeof(_Float16);
  (void)ws_size; (void)n_in; (void)out_size;

  (void)hipMemsetAsync(agg, 0, (size_t)N * GDIM * sizeof(float), stream);
  (void)hipMemsetAsync(deg, 0, (size_t)N * sizeof(unsigned int), stream);

  {
    int n = N * GDIM;
    cvt_f32_to_f16_kernel<<<(n + 255) / 256, 256, 0, stream>>>(z, zh, n);
    cvt_f32_to_f16_kernel<<<(GDIM * GDIM + 255) / 256, 256, 0, stream>>>(W1, w1h, GDIM * GDIM);
    cvt_f32_to_f16_kernel<<<(GDIM * GDIM + 255) / 256, 256, 0, stream>>>(W2, w2h, GDIM * GDIM);
  }

  degree_kernel<<<(E + 255) / 256, 256, 0, stream>>>(src, deg, E);
  alpha_kernel<<<(E + 255) / 256, 256, 0, stream>>>(src, deg, alpha, E);

  const int edgesPerGroup = WPB * EPW;  // 128
  const int nGroups = (E + edgesPerGroup - 1) / edgesPerGroup;
  int blocks = nGroups < 1250 ? nGroups : 1250;   // ~4 groups/block amortize staging
  edge_mlp_kernel<<<blocks, 256, 0, stream>>>(zh, src, dst, alpha, w1h, w2h,
                                              b1, b2, agg, E, nGroups);

  layernorm_kernel<<<(N + WPB - 1) / WPB, 256, 0, stream>>>(z, agg, gamma, beta, out, N);
}
